// DynamicConv_59571196396233
// MI455X (gfx1250) — compile-verified, matrix operand order checked by reading, not run on hardware
//
#include <hip/hip_runtime.h>
#include <hip/hip_bf16.h>

// DynamicConv (involution): B=8, C=192, H=W=56, K=5, P=2, S=1, HEADS=8, CG=24
// Kernel 0: pre-pack filt_w into WMMA-B-fragment order (bf16, zero-padded) in d_ws.
// Kernel 1: fused — per-workgroup WMMA filter-generation GEMM into LDS, then
//           depthwise dynamic-filter application. v_wmma_f32_16x16x32_bf16.

typedef __attribute__((ext_vector_type(16))) __bf16 v16bf;
typedef __attribute__((ext_vector_type(8)))  float  v8f;

#define Bq    8
#define Cq    192
#define Hq    56
#define Wq    56
#define KK    5
#define HEADS 8
#define CG    24
#define NF    200                 // K*K*HEADS
#define HW    (Hq * Wq)           // 3136
#define TILE_ROWS 2
#define TPIX  (TILE_ROWS * Wq)    // 112 pixels per tile (memory-contiguous)
#define MTILES (TPIX / 16)        // 7
#define NTILES 13                 // ceil(200/16)
#define KSTEPS 6                  // 192 / 32
#define FSTRIDE 201               // odd stride -> conflict-free apply reads
#define THREADS 256
#define APPLY_ELEMS (TPIX * Cq)             // 21504
#define APPLY_ITERS (APPLY_ELEMS / THREADS) // 84
#define BPACK_DWORDS (NTILES * KSTEPS * 256)   // 19968 dwords = 78 KB

// ---------------------------------------------------------------------------
// Kernel 0: pack filt_w -> WMMA B-fragment layout (ISA 7.12.2, 16-bit B 32x16,
// transpose-symmetric to the A layout: lane = N column, VGPR v holds the K
// pair  group*16 + half*8 + vv*2 ). One dword = bf16(row c) | bf16(row c+1)<<16.
// Fragment (t,s) occupies 256 consecutive dwords: lane*8 + v.
// ---------------------------------------------------------------------------
__global__ __launch_bounds__(THREADS) void dynconv_prepack_kernel(
    const float* __restrict__ filt_w,   // [C, NF]
    unsigned*    __restrict__ bpack)    // [BPACK_DWORDS]
{
    const int d = blockIdx.x * THREADS + threadIdx.x;   // < BPACK_DWORDS
    const int fi    = d >> 8;           // t*KSTEPS + s
    const int t     = fi / KSTEPS;
    const int s     = fi % KSTEPS;
    const int chunk = d & 255;
    const int lane  = chunk >> 3;
    const int v     = chunk & 7;
    const int n     = 16 * t + (lane & 15);
    const int half  = lane >> 4;
    const int kk    = (v >> 2) * 16 + half * 8 + (v & 3) * 2;
    const int c     = 32 * s + kk;

    float v0 = 0.0f, v1 = 0.0f;
    if (n < NF) {
        v0 = filt_w[(size_t)c * NF + n];
        v1 = filt_w[(size_t)(c + 1) * NF + n];
    }
    union { __bf16 h[2]; unsigned u; } pk;
    pk.h[0] = (__bf16)v0;
    pk.h[1] = (__bf16)v1;
    bpack[d] = pk.u;
}

// ---------------------------------------------------------------------------
// Kernel 1: fused filter-generation (WMMA) + application.
// ---------------------------------------------------------------------------
__global__ __launch_bounds__(THREADS) void dynconv_fused_kernel(
    const float*    __restrict__ x,      // [B, C, H, W]
    const unsigned* __restrict__ bpack,  // pre-packed B fragments
    float*          __restrict__ out)    // [B, C, H, W]
{
    extern __shared__ float fsm[];       // [TPIX][FSTRIDE]

    const int blk = blockIdx.x;          // [0, B * H/TILE_ROWS)
    const int b   = blk / (Hq / TILE_ROWS);
    const int i0  = (blk % (Hq / TILE_ROWS)) * TILE_ROWS;
    const int tid  = threadIdx.x;
    const int lane = tid & 31;
    const int wave = tid >> 5;
    const int m    = lane & 15;
    const int half = lane >> 4;

    // ---------------- Phase 1: filter-generation GEMM (waves 0..6) ----------
    if (wave < MTILES) {
        const int pixBase = 16 * wave;
        const float* xA = x + (size_t)b * Cq * HW + (size_t)i0 * Wq + pixBase + m;

        // Preload A fragments for all 6 K-steps (ISA 7.12.2 16-bit A 16x32).
        // Lane-coalesced dword loads (consecutive pixels across lanes).
        v16bf afrag[KSTEPS];
        #pragma unroll
        for (int s = 0; s < KSTEPS; ++s) {
            #pragma unroll
            for (int v = 0; v < 8; ++v) {
                const int kk = (v >> 2) * 16 + half * 8 + (v & 3) * 2;
                const int c  = 32 * s + kk;
                const float* p = xA + (size_t)c * HW;
                afrag[s][2 * v]     = (__bf16)p[0];
                afrag[s][2 * v + 1] = (__bf16)p[HW];   // channel c+1
            }
        }

        // B fragments: one aligned 32B vector load per WMMA (no predication).
        const v16bf* bfrags = (const v16bf*)bpack;     // 32 v16bf per (t,s)

        for (int t = 0; t < NTILES; ++t) {
            v8f acc = {};
            #pragma unroll
            for (int s = 0; s < KSTEPS; ++s) {
                const v16bf bfrag = bfrags[(size_t)(t * KSTEPS + s) * 32 + lane];
                acc = __builtin_amdgcn_wmma_f32_16x16x32_bf16(
                          false, afrag[s], false, bfrag,
                          (short)0, acc, false, false);
            }
            const int n = 16 * t + m;
            if (n < NF) {
                #pragma unroll
                for (int v = 0; v < 8; ++v) {
                    const int mrow = half * 8 + v;     // f32 C/D layout
                    fsm[(pixBase + mrow) * FSTRIDE + n] = acc[v];
                }
            }
        }
    }

    __syncthreads();

    // ---------------- Phase 2: apply per-pixel dynamic 5x5 filters ----------
    #pragma unroll 2
    for (int it = 0; it < APPLY_ITERS; ++it) {
        const int idx = tid + THREADS * it;   // [0, TPIX*C)
        const int c   = idx / TPIX;
        const int pix = idx % TPIX;           // consecutive lanes -> consecutive j
        const int r   = pix / Wq;
        const int j   = pix % Wq;
        const int i   = i0 + r;
        const int h   = c / CG;

        const float* xc = x + ((size_t)b * Cq + c) * HW;
        const float* fp = fsm + pix * FSTRIDE + h;

        float sum = 0.0f;
        #pragma unroll
        for (int ki = 0; ki < KK; ++ki) {
            const int ii = i + ki - 2;
            if ((unsigned)ii < (unsigned)Hq) {
                #pragma unroll
                for (int kj = 0; kj < KK; ++kj) {
                    const int jj = j + kj - 2;
                    if ((unsigned)jj < (unsigned)Wq) {
                        sum = fmaf(xc[ii * Wq + jj],
                                   fp[ki * (KK * HEADS) + kj * HEADS],
                                   sum);
                    }
                }
            }
        }
        out[((size_t)b * Cq + c) * HW + (size_t)i * Wq + j] = sum;
    }
}

extern "C" void kernel_launch(void* const* d_in, const int* in_sizes, int n_in,
                              void* d_out, int out_size, void* d_ws, size_t ws_size,
                              hipStream_t stream) {
    (void)in_sizes; (void)n_in; (void)ws_size; (void)out_size;
    const float* x      = (const float*)d_in[0];
    const float* filt_w = (const float*)d_in[1];
    float*       out    = (float*)d_out;
    unsigned*    bpack  = (unsigned*)d_ws;             // 78 KB of scratch

    // Kernel 0: pre-pack B fragments (same for all tiles; L2-resident after).
    dynconv_prepack_kernel<<<BPACK_DWORDS / THREADS, THREADS, 0, stream>>>(
        filt_w, bpack);

    // Kernel 1: fused WMMA filter-gen + apply.
    const int grid = Bq * (Hq / TILE_ROWS);            // 224 workgroups
    const size_t lds_bytes = (size_t)TPIX * FSTRIDE * sizeof(float); // ~90 KB
    dynconv_fused_kernel<<<grid, THREADS, lds_bytes, stream>>>(x, bpack, out);
}
